// MatchingField_45870250721616
// MI455X (gfx1250) — matrix operand order repaired and checked by reading, not compiled, not run on hardware
//
#include <hip/hip_runtime.h>
#include <hip/hip_bf16.h>

typedef __attribute__((ext_vector_type(16))) _Float16 v16h;
typedef __attribute__((ext_vector_type(8)))  float    v8f;

#define LOWW 256          // low-res width/height (stage 0, res=2)
#define NRAY (LOWW*LOWW)  // rays per view
#define NS   64           // samples per ray
#define VOLN 128
#define FULLW 512
#define NVIEW 3
#define BLOCKS_PER_VIEW 512   // 65536 rays / (16 rays/wave * 8 waves/block)

__device__ __forceinline__ float trilookup(const float* __restrict__ vol,
                                           float px, float py, float pz) {
    // grid coords in [-1,1] -> [0,127], border clamp, align_corners=True
    float x = fminf(fmaxf((px + 1.0f) * 0.5f * (float)(VOLN - 1), 0.0f), (float)(VOLN - 1));
    float y = fminf(fmaxf((py + 1.0f) * 0.5f * (float)(VOLN - 1), 0.0f), (float)(VOLN - 1));
    float z = fminf(fmaxf((pz + 1.0f) * 0.5f * (float)(VOLN - 1), 0.0f), (float)(VOLN - 1));
    float x0f = floorf(x), y0f = floorf(y), z0f = floorf(z);
    float wx = x - x0f, wy = y - y0f, wz = z - z0f;
    int x0 = (int)x0f, y0 = (int)y0f, z0 = (int)z0f;
    int x1 = min(x0 + 1, VOLN - 1);
    int y1 = min(y0 + 1, VOLN - 1);
    int z1 = min(z0 + 1, VOLN - 1);
    const int P = VOLN, PP = VOLN * VOLN;
    float c000 = vol[z0 * PP + y0 * P + x0];
    float c001 = vol[z0 * PP + y0 * P + x1];
    float c010 = vol[z0 * PP + y1 * P + x0];
    float c011 = vol[z0 * PP + y1 * P + x1];
    float c100 = vol[z1 * PP + y0 * P + x0];
    float c101 = vol[z1 * PP + y0 * P + x1];
    float c110 = vol[z1 * PP + y1 * P + x0];
    float c111 = vol[z1 * PP + y1 * P + x1];
    float c00 = c000 * (1.f - wx) + c001 * wx;
    float c01 = c010 * (1.f - wx) + c011 * wx;
    float c10 = c100 * (1.f - wx) + c101 * wx;
    float c11 = c110 * (1.f - wx) + c111 * wx;
    float c0 = c00 * (1.f - wy) + c01 * wy;
    float c1 = c10 * (1.f - wy) + c11 * wy;
    return c0 * (1.f - wz) + c1 * wz;
}

__global__ void mf_init(float* __restrict__ acc) {
    if (threadIdx.x < 3 * NVIEW) acc[threadIdx.x] = 0.0f;
}

// One wave32 handles a tile of 16 rays x 64 samples.
// Lane l (l<16) owns ray M=l, samples K in {0..7,16..23} (+32 for chunk 1);
// lane l+16 owns ray M=l, samples K in {8..15,24..31} (+32) -- this is exactly
// the 16-bit A-matrix VGPR layout of V_WMMA_F32_16X16X32_F16 (ISA 7.12.2).
__global__ void __launch_bounds__(256)
mf_render(const float* __restrict__ intrs, const float* __restrict__ c2ws,
          const float* __restrict__ nfs, const float* __restrict__ vol,
          float* __restrict__ depth256, float* __restrict__ occAcc) {
    __shared__ float smem[8][32];

    const int lane = threadIdx.x & 31;
    const int wv   = threadIdx.x >> 5;
    const int view = blockIdx.x / BLOCKS_PER_VIEW;
    const int blkInView = blockIdx.x % BLOCKS_PER_VIEW;
    const int waveInView = blkInView * 8 + wv;
    const int n  = lane & 15;             // ray-in-tile / WMMA column id
    const bool hi = lane >= 16;
    const int rid = waveInView * 16 + n;  // ray id within view
    const int pyI = rid >> 8, pxI = rid & (LOWW - 1);

    // pixel coords: linspace(0, 511, 256)
    const float px = (float)pxI * (511.0f / 255.0f);
    const float py = (float)pyI * (511.0f / 255.0f);

    // generic 3x3 intrinsic inverse, cam = inv(K) @ (px,py,1)
    const float* Kc = intrs + view * 9;
    float a = Kc[0], b = Kc[1], c = Kc[2];
    float d_ = Kc[3], e = Kc[4], f = Kc[5];
    float g = Kc[6], h = Kc[7], i = Kc[8];
    float det = a * (e * i - f * h) - b * (d_ * i - f * g) + c * (d_ * h - e * g);
    float idet = 1.0f / det;
    float camx = ((e * i - f * h) * px - (b * i - c * h) * py + (b * f - c * e)) * idet;
    float camy = (-(d_ * i - f * g) * px + (a * i - c * g) * py - (a * f - c * d_)) * idet;
    float camz = ((d_ * h - e * g) * px - (a * h - b * g) * py + (a * e - b * d_)) * idet;
    float inl = rsqrtf(camx * camx + camy * camy + camz * camz);
    float dcx = camx * inl, dcy = camy * inl, dcz = camz * inl;
    float camzn = camz * inl;  // cam_rays_d.z (camera-frame normalized z)

    // world-space ray dir and origin
    const float* M = c2ws + view * 16;
    float dwx = M[0] * dcx + M[1] * dcy + M[2] * dcz;
    float dwy = M[4] * dcx + M[5] * dcy + M[6] * dcz;
    float dwz = M[8] * dcx + M[9] * dcy + M[10] * dcz;
    float tx = M[3], ty = M[7], tz = M[11];

    float znear = nfs[view * 2 + 0];
    float zstep = (nfs[view * 2 + 1] - znear) * (1.0f / (float)(NS - 1));

    float dens[32];
    float pO = 0.0f, pc = 0.0f;
#pragma unroll
    for (int it = 0; it < 32; ++it) {
        int chunk = it >> 4, ee = it & 15;
        int K = ((ee < 8) ? ee : ee + 8) + (hi ? 8 : 0);
        int s = chunk * 32 + K;
        float z = znear + zstep * (float)s;
        float wxp = tx + dwx * z;
        float wyp = ty + dwy * z;
        float wzp = tz + dwz * z;
        float dd = trilookup(vol, wxp, wyp, wzp);
        dens[it] = dd;
        float r2 = wxp * wxp + wyp * wyp + wzp * wzp;
        float o = (r2 > 1.0f) ? 1.0f : 0.0f;
        pO += dd * o;
        pc += o;
    }

    // occ-reg part 1: samples s<6 live in chunk0, e=0..5 of the low half-wave
    float p6 = 0.0f;
    {
        float t6 = dens[0] + dens[1] + dens[2] + dens[3] + dens[4] + dens[5];
        p6 = hi ? 0.0f : t6;
    }

    // per-ray max for stable softmax (lanes l and l+16 share a ray)
    float mx = -1e30f;
#pragma unroll
    for (int it = 0; it < 32; ++it) mx = fmaxf(mx, dens[it]);
    mx = fmaxf(mx, __shfl_xor(mx, 16, 32));

    // A = exp(d - max) (f16), B col0 = ones -> sum(exp), col1 = z -> sum(exp*z)
    v16h a0, a1, b0, b1;
#pragma unroll
    for (int ee = 0; ee < 16; ++ee) {
        a0[ee] = (_Float16)__expf(dens[ee] - mx);
        a1[ee] = (_Float16)__expf(dens[16 + ee] - mx);
        int K = ((ee < 8) ? ee : ee + 8) + (hi ? 8 : 0);  // B: lanes 0-15 hold K={0..7,16..23}, lanes 16-31 K={8..15,24..31}
        float z0v = znear + zstep * (float)(K);
        float z1v = znear + zstep * (float)(32 + K);
        _Float16 e0 = (n == 0) ? (_Float16)1.0f : (n == 1) ? (_Float16)z0v : (_Float16)0.0f;
        _Float16 e1 = (n == 0) ? (_Float16)1.0f : (n == 1) ? (_Float16)z1v : (_Float16)0.0f;
        b0[ee] = e0;
        b1[ee] = e1;
    }

    v8f cacc = {};
    cacc = __builtin_amdgcn_wmma_f32_16x16x32_f16(false, a0, false, b0,
                                                  (short)0, cacc, false, false);
    cacc = __builtin_amdgcn_wmma_f32_16x16x32_f16(false, a1, false, b1,
                                                  (short)0, cacc, false, false);

    // C layout: VGPR r holds rows M=r (lanes 0-15, N=lane) and M=8+r (lanes 16-31).
    // Column 0 (exp sums) in lanes 0/16; column 1 (z-weighted sums) in lanes 1/17.
    if (lane == 0) {
#pragma unroll
        for (int r = 0; r < 8; ++r) smem[wv][r] = cacc[r];
    } else if (lane == 16) {
#pragma unroll
        for (int r = 0; r < 8; ++r) smem[wv][8 + r] = cacc[r];
    } else if (lane == 1) {
#pragma unroll
        for (int r = 0; r < 8; ++r) smem[wv][16 + r] = cacc[r];
    } else if (lane == 17) {
#pragma unroll
        for (int r = 0; r < 8; ++r) smem[wv][24 + r] = cacc[r];
    }
    __syncthreads();

    if (!hi) {
        float es = smem[wv][n];
        float zs = smem[wv][16 + n];
        depth256[view * NRAY + rid] = (zs / es) * camzn;
    }

    // wave-reduce occ partials, one atomic per wave per view
#pragma unroll
    for (int off = 16; off > 0; off >>= 1) {
        p6 += __shfl_xor(p6, off, 32);
        pO += __shfl_xor(pO, off, 32);
        pc += __shfl_xor(pc, off, 32);
    }
    if (lane == 0) {
        atomicAdd(&occAcc[view * 3 + 0], p6);
        atomicAdd(&occAcc[view * 3 + 1], pO);
        atomicAdd(&occAcc[view * 3 + 2], pc);
    }
}

__global__ void mf_finalize(const float* __restrict__ acc, float* __restrict__ out) {
    int v = threadIdx.x;
    if (v < NVIEW) {
        float occ = acc[v * 3 + 0] * (1.0f / ((float)NRAY * 6.0f)) +
                    acc[v * 3 + 1] / (acc[v * 3 + 2] + 1e-10f);
        out[NVIEW * FULLW * FULLW + v] = occ;
    }
}

__global__ void mf_upsample(const float* __restrict__ d256, float* __restrict__ out) {
    int idx = blockIdx.x * 256 + threadIdx.x;
    if (idx >= NVIEW * FULLW * FULLW) return;
    int v = idx / (FULLW * FULLW);
    int rem = idx % (FULLW * FULLW);
    int oy = rem >> 9, ox = rem & (FULLW - 1);
    // half-pixel bilinear (jax.image.resize 'bilinear'), 2x upsample
    float sy = fminf(fmaxf(0.5f * (float)oy - 0.25f, 0.0f), (float)(LOWW - 1));
    float sx = fminf(fmaxf(0.5f * (float)ox - 0.25f, 0.0f), (float)(LOWW - 1));
    int y0 = (int)sy, x0 = (int)sx;
    int y1 = min(y0 + 1, LOWW - 1), x1 = min(x0 + 1, LOWW - 1);
    float wy = sy - (float)y0, wx = sx - (float)x0;
    const float* p = d256 + v * NRAY;
    float v00 = p[y0 * LOWW + x0], v01 = p[y0 * LOWW + x1];
    float v10 = p[y1 * LOWW + x0], v11 = p[y1 * LOWW + x1];
    out[idx] = (v00 * (1.f - wx) + v01 * wx) * (1.f - wy) +
               (v10 * (1.f - wx) + v11 * wx) * wy;
}

extern "C" void kernel_launch(void* const* d_in, const int* in_sizes, int n_in,
                              void* d_out, int out_size, void* d_ws, size_t ws_size,
                              hipStream_t stream) {
    // setup_inputs order: imgs(0, unused), intrs(1), c2ws(2), near_fars(3),
    // density_volume(4), stage_idx(5, hardcoded stage 0)
    const float* intrs = (const float*)d_in[1];
    const float* c2ws  = (const float*)d_in[2];
    const float* nfs   = (const float*)d_in[3];
    const float* vol   = (const float*)d_in[4];
    float* out = (float*)d_out;

    float* depth256 = (float*)d_ws;                 // 3 * 256*256 floats
    float* acc      = (float*)d_ws + NVIEW * NRAY;  // 9 accumulator floats

    mf_init<<<1, 32, 0, stream>>>(acc);
    mf_render<<<NVIEW * BLOCKS_PER_VIEW, 256, 0, stream>>>(intrs, c2ws, nfs, vol,
                                                           depth256, acc);
    mf_finalize<<<1, 32, 0, stream>>>(acc, out);
    mf_upsample<<<(NVIEW * FULLW * FULLW + 255) / 256, 256, 0, stream>>>(depth256, out);
}
